// MambaMixer_90958817394793
// MI455X (gfx1250) — compile-verified, matrix operand order checked by reading, not run on hardware
//
#include <hip/hip_runtime.h>

// Problem dims (fixed by the reference)
#define HD     2048          // H
#define DM     4096
#define NSTATE 16
#define KCONV  4
#define RRANK  128
#define LSEQ   1024
#define PROJW  (2 * DM)              // 8192
#define SSMW   (RRANK + 2 * NSTATE)  // 160

typedef _Float16     v16h __attribute__((ext_vector_type(16)));
typedef float        v8f  __attribute__((ext_vector_type(8)));
typedef unsigned int v4u  __attribute__((ext_vector_type(4)));

union FragU { v16h h; v4u u[2]; };

// Load one 16x32 f16 WMMA operand fragment given this lane's row pointer
// (already offset by lane: row = row0 + (lane&15), col base = (lane>>4)*8).
// Per the CDNA5 ISA 16-bit operand layout this is two contiguous b128 loads:
// halves [k .. k+7] -> VGPRs 0..3 and [k+16 .. k+23] -> VGPRs 4..7.
__device__ __forceinline__ v16h load_frag_at(const _Float16* __restrict__ p,
                                             int elemoff) {
  FragU f;
  f.u[0] = *(const v4u*)(p + elemoff);
  f.u[1] = *(const v4u*)(p + elemoff + 16);
  return f.h;
}

__device__ __forceinline__ v8f wmma32(v16h a, v16h b, v8f c) {
  return __builtin_amdgcn_wmma_f32_16x16x32_f16(false, a, false, b,
                                                (short)0, c, false, false);
}

__device__ __forceinline__ float softplusf(float x) {
  return (x > 20.0f) ? x : log1pf(__expf(x));
}
__device__ __forceinline__ float siluf(float x) {
  return x / (1.0f + __expf(-x));
}

// C[M x N] = A[M x K] @ W[N x K]^T.  A, W f16 row-major (K contiguous), C f32
// row-major (ldc = N). One wave computes a 32(M) x (NSUB*16)(N) tile with
// 2*NSUB WMMAs per 32-wide K step. Branch-free (NSUB is compile-time, grids
// are exact). K is a multiple of 64, so the K loop is unrolled by 2 with
// ping-pong stage buffers (no register rotation copies) and all fragment
// pointers advance by a constant 64 elements per iteration, keeping loads at
// immediate offsets {0,16,32,48,64,80} halves.
// EPI: 0 = plain store, 1 = softplus(x + bias[col]).
template <int EPI, int NSUB>
__global__ void gemm_wmma_kernel(const _Float16* __restrict__ A,
                                 const _Float16* __restrict__ W,
                                 float* __restrict__ C,
                                 const float* __restrict__ bias,
                                 int N, int K, int ntiles, int n_base) {
  const int lane = threadIdx.x & 31;
  const int gw   = blockIdx.x * (blockDim.x >> 5) + (threadIdx.x >> 5);
  const int m0 = (gw / ntiles) * 32;
  const int n0 = n_base + (gw % ntiles) * (NSUB * 16);

  const int rl = lane & 15;            // row-within-tile lane
  const int kl = (lane >> 4) << 3;     // +8 halves for upper half-wave

  const _Float16* ap0 = A + (size_t)(m0 + rl) * K + kl;
  const _Float16* ap1 = ap0 + (size_t)16 * K;
  const _Float16* bp[NSUB];
#pragma unroll
  for (int i = 0; i < NSUB; ++i)
    bp[i] = W + (size_t)(n0 + 16 * i + rl) * K + kl;

  v8f acc[2][NSUB] = {};
  v16h a0[2], a1[2], b[2][NSUB];

  // preload stage 0 (K step 0)
  a0[0] = load_frag_at(ap0, 0);
  a1[0] = load_frag_at(ap1, 0);
#pragma unroll
  for (int i = 0; i < NSUB; ++i) b[0][i] = load_frag_at(bp[i], 0);

  for (int kb = 64; kb < K; kb += 64) {
    // load stage 1 (current base + 32), issue WMMAs for stage 0
    a0[1] = load_frag_at(ap0, 32);
    a1[1] = load_frag_at(ap1, 32);
#pragma unroll
    for (int i = 0; i < NSUB; ++i) b[1][i] = load_frag_at(bp[i], 32);
#pragma unroll
    for (int i = 0; i < NSUB; ++i) {
      acc[0][i] = wmma32(a0[0], b[0][i], acc[0][i]);
      acc[1][i] = wmma32(a1[0], b[0][i], acc[1][i]);
    }
    // load stage 0 (current base + 64), issue WMMAs for stage 1
    a0[0] = load_frag_at(ap0, 64);
    a1[0] = load_frag_at(ap1, 64);
#pragma unroll
    for (int i = 0; i < NSUB; ++i) b[0][i] = load_frag_at(bp[i], 64);
#pragma unroll
    for (int i = 0; i < NSUB; ++i) {
      acc[0][i] = wmma32(a0[1], b[1][i], acc[0][i]);
      acc[1][i] = wmma32(a1[1], b[1][i], acc[1][i]);
    }
    ap0 += 64; ap1 += 64;
#pragma unroll
    for (int i = 0; i < NSUB; ++i) bp[i] += 64;
  }

  // tail: stages S-2 (in buffer 0) and S-1 (load now)
  a0[1] = load_frag_at(ap0, 32);
  a1[1] = load_frag_at(ap1, 32);
#pragma unroll
  for (int i = 0; i < NSUB; ++i) b[1][i] = load_frag_at(bp[i], 32);
#pragma unroll
  for (int i = 0; i < NSUB; ++i) {
    acc[0][i] = wmma32(a0[0], b[0][i], acc[0][i]);
    acc[1][i] = wmma32(a1[0], b[0][i], acc[1][i]);
  }
#pragma unroll
  for (int i = 0; i < NSUB; ++i) {
    acc[0][i] = wmma32(a0[1], b[1][i], acc[0][i]);
    acc[1][i] = wmma32(a1[1], b[1][i], acc[1][i]);
  }

  // C/D layout: VGPR r -> M = m0 + mi*16 + r (+8 for upper half-wave), N = lane&15.
  const int rb = (lane >> 4) << 3;
#pragma unroll
  for (int mi = 0; mi < 2; ++mi) {
#pragma unroll
    for (int i = 0; i < NSUB; ++i) {
      const int col = n0 + 16 * i + rl;
#pragma unroll
      for (int r = 0; r < 8; ++r) {
        float v = acc[mi][i][r];
        if (EPI == 1) v = softplusf(v + bias[col]);
        C[(size_t)(m0 + mi * 16 + rb + r) * N + col] = v;
      }
    }
  }
}

__global__ void cvt_f16_kernel(const float* __restrict__ s,
                               _Float16* __restrict__ d, int n) {
  int i = blockIdx.x * blockDim.x + threadIdx.x;
  if (i < n) d[i] = (_Float16)s[i];
}

// Depthwise causal conv (K=4) + bias + SiLU on h = proj[:, :DM].
__global__ void conv_silu_kernel(const float* __restrict__ proj,
                                 const float* __restrict__ cw,
                                 const float* __restrict__ cb,
                                 float* __restrict__ hc32,
                                 _Float16* __restrict__ hc16) {
  int idx = blockIdx.x * blockDim.x + threadIdx.x;
  if (idx >= LSEQ * DM) return;
  int d = idx & (DM - 1);
  int t = idx >> 12;   // DM == 4096
  float s = cb[d];
#pragma unroll
  for (int j = 0; j < KCONV; ++j) {
    int tt = t - (KCONV - 1) + j;
    if (tt >= 0) s = fmaf(proj[(size_t)tt * PROJW + d], cw[d * KCONV + j], s);
  }
  s = siluf(s);
  hc32[idx] = s;
  hc16[idx] = (_Float16)s;
}

// Extract ts = ssm_p[:, :R] as f16 for the dt GEMM.
__global__ void ts_cvt_kernel(const float* __restrict__ ssm,
                              _Float16* __restrict__ ts16) {
  int i = blockIdx.x * blockDim.x + threadIdx.x;
  if (i >= LSEQ * RRANK) return;
  int t = i >> 7;               // RRANK == 128
  int c = i & (RRANK - 1);
  ts16[i] = (_Float16)ssm[t * SSMW + c];
}

// Selective scan. 4 lanes cooperate per channel d (4 states each); y reduced
// with shfl_xor. Fuses +hc*D, *silu(gate), and f16 output for the final GEMM.
__global__ void scan_kernel(const float* __restrict__ dt,
                            const float* __restrict__ ssm,
                            const float* __restrict__ hc32,
                            const float* __restrict__ proj,
                            const float* __restrict__ A_log,
                            const float* __restrict__ D_param,
                            _Float16* __restrict__ ys16) {
  int gid = blockIdx.x * blockDim.x + threadIdx.x;   // DM * 4 threads
  int d   = gid >> 2;
  int sub = gid & 3;
  int nb  = sub * 4;
  float a[4];
#pragma unroll
  for (int n = 0; n < 4; ++n) a[n] = -__expf(A_log[d * NSTATE + nb + n]);
  float st[4] = {0.f, 0.f, 0.f, 0.f};
  const float Dp = D_param[d];
  for (int t = 0; t < LSEQ; ++t) {
    const float dtv = dt[(size_t)t * DM + d];
    const float hcv = hc32[(size_t)t * DM + d];
    const float* bc = ssm + t * SSMW + RRANK;   // Bm at +0, Cm at +NSTATE
    float y = 0.f;
#pragma unroll
    for (int n = 0; n < 4; ++n) {
      float dA = __expf(dtv * a[n]);
      st[n] = fmaf(dA, st[n], dtv * bc[nb + n] * hcv);
      y = fmaf(st[n], bc[NSTATE + nb + n], y);
    }
    y += __shfl_xor(y, 1, 32);
    y += __shfl_xor(y, 2, 32);
    if (sub == 0) {
      y = fmaf(hcv, Dp, y);
      float g = proj[(size_t)t * PROJW + DM + d];
      y *= siluf(g);
      ys16[(size_t)t * DM + d] = (_Float16)y;
    }
  }
}

extern "C" void kernel_launch(void* const* d_in, const int* in_sizes, int n_in,
                              void* d_out, int out_size, void* d_ws, size_t ws_size,
                              hipStream_t stream) {
  (void)in_sizes; (void)n_in; (void)out_size; (void)ws_size;
  const float* hs     = (const float*)d_in[0];
  const float* W_in   = (const float*)d_in[1];
  const float* conv_w = (const float*)d_in[2];
  const float* conv_b = (const float*)d_in[3];
  const float* W_x    = (const float*)d_in[4];
  const float* W_dt   = (const float*)d_in[5];
  const float* b_dt   = (const float*)d_in[6];
  const float* A_log  = (const float*)d_in[7];
  const float* D_par  = (const float*)d_in[8];
  const float* W_out  = (const float*)d_in[9];

  char* ws = (char*)d_ws;
  size_t off = 0;
  auto carve = [&](size_t bytes) -> void* {
    void* p = ws + off;
    off += (bytes + 255) & ~(size_t)255;
    return p;
  };
  _Float16* hs16   = (_Float16*)carve((size_t)LSEQ * HD * 2);
  _Float16* Win16  = (_Float16*)carve((size_t)PROJW * HD * 2);
  _Float16* Wx16   = (_Float16*)carve((size_t)SSMW * DM * 2);
  _Float16* Wdt16  = (_Float16*)carve((size_t)DM * RRANK * 2);
  _Float16* Wout16 = (_Float16*)carve((size_t)HD * DM * 2);
  float*    proj   = (float*)carve((size_t)LSEQ * PROJW * 4);
  float*    hc32   = (float*)carve((size_t)LSEQ * DM * 4);
  _Float16* hc16   = (_Float16*)carve((size_t)LSEQ * DM * 2);
  float*    ssm    = (float*)carve((size_t)LSEQ * SSMW * 4);
  _Float16* ts16   = (_Float16*)carve((size_t)LSEQ * RRANK * 2);
  float*    dtb    = (float*)carve((size_t)LSEQ * DM * 4);
  _Float16* ys16   = (_Float16*)carve((size_t)LSEQ * DM * 2);

  auto cvt = [&](const float* s, _Float16* d, int n) {
    cvt_f16_kernel<<<(n + 255) / 256, 256, 0, stream>>>(s, d, n);
  };
  cvt(hs,    hs16,   LSEQ * HD);
  cvt(W_in,  Win16,  PROJW * HD);
  cvt(W_x,   Wx16,   SSMW * DM);
  cvt(W_dt,  Wdt16,  DM * RRANK);
  cvt(W_out, Wout16, HD * DM);

  // tiles = (M/32) * ntiles is always a multiple of 8 (8 waves / 256-thr block)
  auto grid_for = [](int ntiles) { return ((LSEQ / 32) * ntiles) / 8; };

  // 1) proj = hs @ W_in^T            (1024 x 8192, K=2048)
  gemm_wmma_kernel<0, 4><<<grid_for(128), 256, 0, stream>>>(
      hs16, Win16, proj, nullptr, PROJW, HD, 128, 0);
  // 2) depthwise causal conv + SiLU
  conv_silu_kernel<<<(LSEQ * DM + 255) / 256, 256, 0, stream>>>(
      proj, conv_w, conv_b, hc32, hc16);
  // 3) ssm_p = hc @ W_x^T            (1024 x 160, K=4096): 2 full 64-wide
  //    tiles + one 32-wide edge strip (NSUB=2) at n_base=128.
  gemm_wmma_kernel<0, 4><<<grid_for(2), 256, 0, stream>>>(
      hc16, Wx16, ssm, nullptr, SSMW, DM, 2, 0);
  gemm_wmma_kernel<0, 2><<<grid_for(1), 256, 0, stream>>>(
      hc16, Wx16, ssm, nullptr, SSMW, DM, 1, 128);
  ts_cvt_kernel<<<(LSEQ * RRANK + 255) / 256, 256, 0, stream>>>(ssm, ts16);
  // 4) dt = softplus(ts @ W_dt^T + b_dt)   (1024 x 4096, K=128)
  gemm_wmma_kernel<1, 4><<<grid_for(64), 256, 0, stream>>>(
      ts16, Wdt16, dtb, b_dt, DM, RRANK, 64, 0);
  // 5) selective scan (fuses D-skip + gate)
  scan_kernel<<<(DM * 4 + 255) / 256, 256, 0, stream>>>(
      dtb, ssm, hc32, proj, A_log, D_par, ys16);
  // 6) out = ys @ W_out^T            (1024 x 2048, K=4096)
  gemm_wmma_kernel<0, 4><<<grid_for(32), 256, 0, stream>>>(
      ys16, Wout16, (float*)d_out, nullptr, HD, DM, 32, 0);
}
